// QLSTM_65481071399968
// MI455X (gfx1250) — matrix operand order repaired and checked by reading, not compiled, not run on hardware
//
#include <hip/hip_runtime.h>
#include <hip/hip_bf16.h>

// ---------------------------------------------------------------------------
// QLSTM on gfx1250 (MI455X): bf16 WMMA everywhere, f32 accumulate.
//   SEQ=512, BATCH=128, IN_DIM=1024, HID=1024, GATES=4*HID=4096
// ---------------------------------------------------------------------------

typedef __bf16 bf16_t;
typedef __attribute__((ext_vector_type(16))) __bf16 v16bf;
typedef __attribute__((ext_vector_type(8)))  __bf16 v8bf;
typedef __attribute__((ext_vector_type(8)))  float  v8f;
typedef __attribute__((ext_vector_type(4)))  float  v4f;

#define SEQ     512
#define BATCH   128
#define IN_DIM  1024
#define HID     1024
#define GATES   4096   // 4*HID

__device__ __forceinline__ v8f wmma_bf16(v16bf a, v16bf b, v8f c) {
    return __builtin_amdgcn_wmma_f32_16x16x32_bf16(
        /*neg_a=*/false, a, /*neg_b=*/false, b,
        /*c_mod=*/(short)0, c, /*reuse_a=*/false, /*reuse_b=*/false);
}

// A fragment: 16x32 (MxK) bf16, A row-major with leading dim ld.
// Lane l<16: row=l, K {k0..k0+7, k0+16..k0+23}; lane>=16: row=l-16, +8 shift.
__device__ __forceinline__ v16bf load_a_bf16(const bf16_t* __restrict__ a,
                                             int ld, int k0, int lane) {
    int m  = lane & 15;
    int ko = k0 + ((lane >> 4) << 3);
    const bf16_t* p = a + (size_t)m * ld + ko;
    union { v16bf v; uint4 q[2]; } u;
    u.q[0] = *(const uint4*)(p);
    u.q[1] = *(const uint4*)(p + 16);
    return u.v;
}

// Same A fragment but source is f32; convert to bf16 in-register.
__device__ __forceinline__ v16bf load_a_f32(const float* __restrict__ a,
                                            int ld, int k0, int lane) {
    int m  = lane & 15;
    int ko = k0 + ((lane >> 4) << 3);
    const float* p0 = a + (size_t)m * ld + ko;   // K ko..ko+7
    const float* p1 = p0 + 16;                   // K ko+16..ko+23
    v4f a0 = *(const v4f*)(p0);
    v4f a1 = *(const v4f*)(p0 + 4);
    v4f b0 = *(const v4f*)(p1);
    v4f b1 = *(const v4f*)(p1 + 4);
    v16bf f;
#pragma unroll
    for (int i = 0; i < 4; ++i) {
        f[i]      = (bf16_t)a0[i];
        f[4 + i]  = (bf16_t)a1[i];
        f[8 + i]  = (bf16_t)b0[i];
        f[12 + i] = (bf16_t)b1[i];
    }
    return f;
}

// B fragment: 32x16 (KxN) bf16, sourced from Bt (N x K row-major), ld = K stride.
// Lane l<16: col=l, K {k0..k0+7, k0+16..k0+23}; lane>=16: col=l-16, +8 shift.
__device__ __forceinline__ v16bf load_b_bf16(const bf16_t* __restrict__ bt,
                                             int ld, int k0, int lane) {
    int n  = lane & 15;
    int ko = k0 + ((lane >> 4) << 3);
    const bf16_t* p = bt + (size_t)n * ld + ko;
    union { v16bf v; uint4 q[2]; } u;
    u.q[0] = *(const uint4*)(p);
    u.q[1] = *(const uint4*)(p + 16);
    return u.v;
}

__device__ __forceinline__ float sigmoidf_(float x) {
    return 1.0f / (1.0f + __expf(-x));
}

// ---------------------------------------------------------------------------
// Prep kernels
// ---------------------------------------------------------------------------
__global__ void cvt_f32_bf16_kernel(const float* __restrict__ src,
                                    bf16_t* __restrict__ dst, int n) {
    int i = blockIdx.x * blockDim.x + threadIdx.x;
    if (i < n) dst[i] = (bf16_t)src[i];
}

__global__ void init_state_kernel(bf16_t* __restrict__ hx_bf,
                                  float* __restrict__ cx, int n) {
    int i = blockIdx.x * blockDim.x + threadIdx.x;
    if (i < n) { hx_bf[i] = (bf16_t)0.0f; cx[i] = 0.0f; }
}

// ---------------------------------------------------------------------------
// x_proj = inputs @ Wx^T + b, stored in WMMA-C fragment-major bf16 layout:
//   xproj[tm][tn][lane*8 + r]   tm in [0,4096) (rows/16), tn in [0,256) (cols/16)
// Grid: (256, 256), block 256 (8 waves). Each wave computes TWO 16x16 C tiles
// (adjacent M strips) per B fragment -> 2 WMMAs per B load.
// ---------------------------------------------------------------------------
__global__ __launch_bounds__(256)
void xproj_kernel(const float* __restrict__ x,       // [SEQ*BATCH, IN_DIM] f32
                  const bf16_t* __restrict__ Wbf,    // [GATES, IN_DIM+HID] bf16
                  const float* __restrict__ bias,    // [GATES]
                  bf16_t* __restrict__ xproj) {
    const int lane = threadIdx.x & 31;
    const int wave = threadIdx.x >> 5;
    const int tn   = blockIdx.x;                      // 0..255
    const int tm0  = blockIdx.y * 16 + wave * 2;      // two M tiles per wave
    const int n0   = tn * 16;

    const float*  arow0 = x + (size_t)(tm0 * 16) * IN_DIM;
    const float*  arow1 = arow0 + (size_t)16 * IN_DIM;
    const bf16_t* brow  = Wbf + (size_t)n0 * (IN_DIM + HID);   // Wx cols [0,1024)

    v8f acc0 = {}, acc1 = {};
#pragma unroll 4
    for (int k = 0; k < IN_DIM; k += 32) {
        v16bf b  = load_b_bf16(brow, IN_DIM + HID, k, lane);
        v16bf a0 = load_a_f32(arow0, IN_DIM, k, lane);
        v16bf a1 = load_a_f32(arow1, IN_DIM, k, lane);
        acc0 = wmma_bf16(a0, b, acc0);
        acc1 = wmma_bf16(a1, b, acc1);
    }

    float bv = bias[n0 + (lane & 15)];
#pragma unroll
    for (int t = 0; t < 2; ++t) {
        union { v8bf v; uint4 q; } st;
        v8f* acc = t ? &acc1 : &acc0;
#pragma unroll
        for (int r = 0; r < 8; ++r) st.v[r] = (bf16_t)((*acc)[r] + bv);
        bf16_t* dst = xproj + (((size_t)(tm0 + t) * 256 + tn) * 256) + lane * 8;
        *(uint4*)dst = st.q;
    }
}

// ---------------------------------------------------------------------------
// Per-step gates: z = x_proj[s] + hx @ Wh^T  (z stored bf16 row-major 128x4096)
// Block 128 (4 waves); each wave computes TWO M strips per B fragment.
// Writes go through an LDS 16x16 tile transpose (ds_store_b16 scatter ->
// s_wait_dscnt -> coalesced b128 stores) so z is row-major for the cell
// kernel's A-fragment loads.
// ---------------------------------------------------------------------------
__global__ __launch_bounds__(128)
void gates_kernel(const bf16_t* __restrict__ hx_bf,  // [BATCH, HID]
                  const bf16_t* __restrict__ Wbf,    // [GATES, IN_DIM+HID]
                  const bf16_t* __restrict__ xproj,  // fragment-major
                  bf16_t* __restrict__ z,            // [BATCH, GATES]
                  int s) {
    __shared__ __align__(16) bf16_t ldsz[4][2][256];  // 2 tiles per wave

    const int lane = threadIdx.x & 31;
    const int wave = threadIdx.x >> 5;            // 0..3
    const int tn   = blockIdx.x;                  // 0..255
    const int m0   = wave * 32;                   // two 16-row strips
    const int n0   = tn * 16;

    const bf16_t* arow0 = hx_bf + (size_t)m0 * HID;
    const bf16_t* arow1 = arow0 + (size_t)16 * HID;
    const bf16_t* brow  = Wbf + (size_t)n0 * (IN_DIM + HID) + IN_DIM;  // Wh part

    v8f acc0 = {}, acc1 = {};
#pragma unroll 4
    for (int k = 0; k < HID; k += 32) {
        v16bf b  = load_b_bf16(brow, IN_DIM + HID, k, lane);
        v16bf a0 = load_a_bf16(arow0, HID, k, lane);
        v16bf a1 = load_a_bf16(arow1, HID, k, lane);
        acc0 = wmma_bf16(a0, b, acc0);
        acc1 = wmma_bf16(a1, b, acc1);
    }

    const int mrow = (lane >> 4) << 3;
#pragma unroll
    for (int t = 0; t < 2; ++t) {
        // add precomputed x_proj fragment (already includes bias)
        const int tm = s * 8 + wave * 2 + t;
        union { v8bf v; uint4 q; } xa;
        xa.q = *(const uint4*)(xproj + (((size_t)tm * 256 + tn) * 256) + lane * 8);
        v8f* acc = t ? &acc1 : &acc0;
#pragma unroll
        for (int r = 0; r < 8; ++r) {
            float v = (*acc)[r] + (float)xa.v[r];
            ldsz[wave][t][(r + mrow) * 16 + (lane & 15)] = (bf16_t)v;
        }
    }
    asm volatile("s_wait_dscnt 0x0" ::: "memory");   // intra-wave LDS RAW

    const int row = lane >> 1, half = lane & 1;
#pragma unroll
    for (int t = 0; t < 2; ++t) {
        uint4 oq = *(const uint4*)&ldsz[wave][t][row * 16 + half * 8];
        *(uint4*)(z + (size_t)(m0 + t * 16 + row) * GATES + n0 + half * 8) = oq;
    }
}

// ---------------------------------------------------------------------------
// Per-step fused qgate + cell update. Each wave computes one 16x16 (m,n) patch
// for ALL FOUR gates (B fragment of qgate_W shared across the 4 WMMAs per
// K-chunk), then does the double-sigmoid/tanh LSTM update in-register.
// Grid: (8, 64), block 32 (one wave per block).
// ---------------------------------------------------------------------------
__global__ __launch_bounds__(32)
void cell_kernel(const bf16_t* __restrict__ z,     // [BATCH, GATES]
                 const bf16_t* __restrict__ Qbf,   // [HID, HID] (row n contig in k)
                 float* __restrict__ cx,           // [BATCH, HID]
                 bf16_t* __restrict__ hx_bf,       // [BATCH, HID]
                 float* __restrict__ outs,         // [SEQ, BATCH, HID] (d_out)
                 float* __restrict__ hx_out,
                 float* __restrict__ cx_out,
                 int s, int last) {
    const int lane = threadIdx.x;
    const int m0   = blockIdx.x * 16;   // batch strip
    const int n0   = blockIdx.y * 16;   // hidden cols

    const bf16_t* zrow = z   + (size_t)m0 * GATES;
    const bf16_t* brow = Qbf + (size_t)n0 * HID;

    v8f af = {}, ai = {}, ag = {}, ao = {};
#pragma unroll 2
    for (int k = 0; k < HID; k += 32) {
        v16bf b  = load_b_bf16(brow, HID, k, lane);
        v16bf a0 = load_a_bf16(zrow + 0 * HID, GATES, k, lane);
        v16bf a1 = load_a_bf16(zrow + 1 * HID, GATES, k, lane);
        v16bf a2 = load_a_bf16(zrow + 2 * HID, GATES, k, lane);
        v16bf a3 = load_a_bf16(zrow + 3 * HID, GATES, k, lane);
        af = wmma_bf16(a0, b, af);
        ai = wmma_bf16(a1, b, ai);
        ag = wmma_bf16(a2, b, ag);
        ao = wmma_bf16(a3, b, ao);
    }

    const int mrow = (lane >> 4) << 3;
    const int n    = n0 + (lane & 15);
#pragma unroll
    for (int r = 0; r < 8; ++r) {
        const int m = m0 + r + mrow;
        const size_t idx = (size_t)m * HID + n;
        float f = sigmoidf_(sigmoidf_(af[r]));
        float i = sigmoidf_(sigmoidf_(ai[r]));
        float g = tanhf   (sigmoidf_(ag[r]));
        float o = sigmoidf_(sigmoidf_(ao[r]));
        float c = f * cx[idx] + i * g;
        float h = o * tanhf(c);
        cx[idx]    = c;
        hx_bf[idx] = (bf16_t)h;
        outs[(size_t)s * (BATCH * HID) + idx] = h;
        if (last) { hx_out[idx] = h; cx_out[idx] = c; }
    }
}

// ---------------------------------------------------------------------------
// Host launcher
// ---------------------------------------------------------------------------
extern "C" void kernel_launch(void* const* d_in, const int* in_sizes, int n_in,
                              void* d_out, int out_size, void* d_ws, size_t ws_size,
                              hipStream_t stream) {
    const float* x      = (const float*)d_in[0];   // [SEQ, BATCH, IN_DIM]
    const float* proj_W = (const float*)d_in[1];   // [GATES, IN_DIM+HID]
    const float* proj_b = (const float*)d_in[2];   // [GATES]
    const float* qgateW = (const float*)d_in[3];   // [HID, HID]

    float* out = (float*)d_out;
    float* outs   = out;                                  // SEQ*BATCH*HID
    float* hx_out = out + (size_t)SEQ * BATCH * HID;      // BATCH*HID
    float* cx_out = hx_out + (size_t)BATCH * HID;         // BATCH*HID

    // workspace layout (bytes)
    char* ws = (char*)d_ws;
    const size_t NW  = (size_t)GATES * (IN_DIM + HID);    // 8,388,608
    const size_t NQ  = (size_t)HID * HID;                 // 1,048,576
    const size_t NXP = (size_t)SEQ * BATCH * GATES;       // 268,435,456
    const size_t NZ  = (size_t)BATCH * GATES;             // 524,288
    const size_t NH  = (size_t)BATCH * HID;               // 131,072

    bf16_t* Wbf   = (bf16_t*)(ws);
    bf16_t* Qbf   = (bf16_t*)(ws + NW * 2);
    bf16_t* xproj = (bf16_t*)(ws + NW * 2 + NQ * 2);
    bf16_t* zbuf  = (bf16_t*)(ws + NW * 2 + NQ * 2 + NXP * 2);
    bf16_t* hx_bf = (bf16_t*)(ws + NW * 2 + NQ * 2 + NXP * 2 + NZ * 2);
    float*  cx    = (float*) (ws + NW * 2 + NQ * 2 + NXP * 2 + NZ * 2 + NH * 2);

    // 1) weight conversions to bf16
    cvt_f32_bf16_kernel<<<(int)((NW + 255) / 256), 256, 0, stream>>>(proj_W, Wbf, (int)NW);
    cvt_f32_bf16_kernel<<<(int)((NQ + 255) / 256), 256, 0, stream>>>(qgateW, Qbf, (int)NQ);

    // 2) zero-init recurrent state (deterministic per call)
    init_state_kernel<<<(int)((NH + 255) / 256), 256, 0, stream>>>(hx_bf, cx, (int)NH);

    // 3) big parallel input projection (+bias), fragment-major bf16 out
    xproj_kernel<<<dim3(GATES / 16, (SEQ * BATCH) / 256), 256, 0, stream>>>(
        x, Wbf, proj_b, xproj);

    // 4) recurrence: 512 serialized steps of (gates GEMM, fused qgate+cell)
    for (int s = 0; s < SEQ; ++s) {
        gates_kernel<<<dim3(GATES / 16), 128, 0, stream>>>(hx_bf, Wbf, xproj, zbuf, s);
        cell_kernel<<<dim3(BATCH / 16, HID / 16), 32, 0, stream>>>(
            zbuf, Qbf, cx, hx_bf, outs, hx_out, cx_out, s, (s == SEQ - 1) ? 1 : 0);
    }
}
